// DeformableSVDModulatedConv2d_74543452390135
// MI455X (gfx1250) — matrix operand order repaired and checked by reading, not compiled, not run on hardware
//
#include <hip/hip_runtime.h>
#include <math.h>

typedef __bf16 bf16_t;
typedef __attribute__((ext_vector_type(16))) __bf16 v16bf;
typedef __attribute__((ext_vector_type(8)))  __bf16 v8bf;
typedef __attribute__((ext_vector_type(8)))  float  v8f;
typedef __attribute__((ext_vector_type(4)))  unsigned int u32x4;
typedef __attribute__((ext_vector_type(8)))  int i32x8;
typedef __attribute__((ext_vector_type(4)))  int i32x4;

#define B_    16
#define CIN   512
#define COUT  512
#define HW_   1024        // 32*32
#define R_    512
#define MDIM  4608        // K*K*CIN
#define SCALE_F 0.014731391274719738f   // 1/sqrt(4608)

// padded LDS row: 512 bf16 data + 4 dwords (8 bf16) TDM pad -> bank rotation
#define LROW  520

union AFrag { v16bf v; v8bf h[2]; };

// ---------------------------------------------------------------------------
// TDM: 1-D tile (nelem bf16 elements, contiguous in memory) -> LDS, with
// padding of 4 dwords inserted every 256 dwords (i.e. every 512 bf16 = one
// logical row) so LDS rows are LROW elements apart.
// D# per CDNA5 ISA ch.8: group0 {count,lds,global,type=2}; group1 {dsize,pad,
// dims,strides}; groups 2/3 (+extra group on this toolchain) zero (<=2D).
// ---------------------------------------------------------------------------
__device__ __forceinline__ unsigned lds_off(const void* p) {
  return (unsigned)(unsigned long long)(uintptr_t)p;
}

__device__ __forceinline__ void tdm_load_1d_padded(unsigned ldsaddr,
                                                   const bf16_t* gptr,
                                                   unsigned nelem) {
  unsigned long long g = (unsigned long long)(uintptr_t)gptr;
  u32x4 g0;
  g0.x = 1u;                                             // count=1 (valid D#)
  g0.y = ldsaddr;                                        // lds_addr
  g0.z = (unsigned)(g & 0xFFFFFFFFull);                  // global_addr lo
  g0.w = (unsigned)((g >> 32) & 0x01FFFFFFull) | (2u << 30); // hi + type=2
  union { i32x8 v; unsigned short h[16]; } g1;
#pragma unroll
  for (int i = 0; i < 8; ++i) g1.v[i] = 0;
  // word0[31:16]: data_size=1(2B) | pad_enable<<4 | pad_interval(7=256dw)<<6
  //               | pad_amount(3 = 4dw)<<9
  g1.h[1]  = (unsigned short)(0x0001u | (1u << 4) | (7u << 6) | (3u << 9));
  g1.h[3]  = (unsigned short)(nelem & 0xFFFFu);          // tensor_dim0 lo
  g1.h[4]  = (unsigned short)(nelem >> 16);              // tensor_dim0 hi
  g1.h[5]  = 1;                                          // tensor_dim1 = 1
  g1.h[7]  = (unsigned short)nelem;                      // tile_dim0 (<=65535)
  g1.h[8]  = 1;                                          // tile_dim1 = 1
  g1.h[10] = (unsigned short)(nelem & 0xFFFFu);          // dim0 stride (unused)
  g1.h[11] = (unsigned short)(nelem >> 16);
  i32x4 z4 = {0, 0, 0, 0};
  i32x8 z8v = {0, 0, 0, 0, 0, 0, 0, 0};
  __builtin_amdgcn_tensor_load_to_lds(g0, g1.v, z4, z4, z8v, 0);
}

// ---------------------------------------------------------------------------
// K1: s[b][cin] = style[b,:] . modulation_w[cin,:] + modulation_b[cin]
// ---------------------------------------------------------------------------
__global__ void k_modulation(const float* __restrict__ style,
                             const float* __restrict__ mw,
                             const float* __restrict__ mb,
                             float* __restrict__ s,
                             float* __restrict__ sumsq) {
  __shared__ float sty[R_];
  int b = blockIdx.x;
  for (int i = threadIdx.x; i < R_; i += blockDim.x) sty[i] = style[b * R_ + i];
  if (threadIdx.x == 0) sumsq[b] = 0.0f;
  __syncthreads();
  for (int cin = threadIdx.x; cin < CIN; cin += blockDim.x) {
    const float* wr = mw + (size_t)cin * R_;
    float acc = 0.f;
    for (int k = 0; k < R_; ++k) acc += sty[k] * wr[k];
    s[b * CIN + cin] = acc + mb[cin];
  }
}

// ---------------------------------------------------------------------------
// K2a: a_bf[b][m][r] = bf16( u[m][r] * dir_delta[batch_directions[b]][r] )
// ---------------------------------------------------------------------------
__global__ void k_pack_a(const float* __restrict__ u,
                         const float* __restrict__ dir_delta,
                         const int* __restrict__ bdir,
                         bf16_t* __restrict__ a_bf) {
  size_t idx = (size_t)blockIdx.x * 256 + threadIdx.x;   // B*MDIM*R
  int rr = (int)(idx & (R_ - 1));
  size_t t = idx >> 9;
  int m = (int)(t % MDIM);
  int b = (int)(t / MDIM);
  float ev = dir_delta[(size_t)bdir[b] * R_ + rr];
  a_bf[idx] = (bf16_t)(u[(size_t)m * R_ + rr] * ev);
}

// ---------------------------------------------------------------------------
// K2b: vhT[n][r] = bf16( vh[r][n] )
// ---------------------------------------------------------------------------
__global__ void k_pack_vhT(const float* __restrict__ vh, bf16_t* __restrict__ vhT) {
  int idx = blockIdx.x * 256 + threadIdx.x;   // COUT*R
  int rr = idx & (R_ - 1);
  int n  = idx >> 9;
  vhT[idx] = (bf16_t)vh[(size_t)rr * COUT + n];
}

// ---------------------------------------------------------------------------
// K2c: xT[b][h][w][cin] = bf16( x[b][cin][h][w] )
// ---------------------------------------------------------------------------
__global__ void k_pack_xT(const float* __restrict__ x, bf16_t* __restrict__ xT) {
  size_t idx = (size_t)blockIdx.x * 256 + threadIdx.x;   // B*HW*CIN
  int cin = (int)(idx & (CIN - 1));
  size_t t = idx >> 9;
  int hw = (int)(t & (HW_ - 1));
  int b  = (int)(t >> 10);
  xT[idx] = (bf16_t)x[((size_t)b * CIN + cin) * HW_ + hw];
}

// ---------------------------------------------------------------------------
// K3: delta[b] = A[b](4608x512) @ vhT^T(512x512), bf16 in / f32 acc.
//     Block: 8 waves sharing one (b, nt) -> 64-row B panel TDM-staged into
//     padded LDS once; each wave owns one 16x64 output tile; A fragments
//     double-buffered from global. Wave-reduced atomic sum(delta^2).
// ---------------------------------------------------------------------------
__global__ void __launch_bounds__(256) k_delta_gemm(
    const bf16_t* __restrict__ a_bf,
    const bf16_t* __restrict__ vhT,
    bf16_t* __restrict__ delta_bf,
    float* __restrict__ sumsq) {
  __shared__ __align__(16) bf16_t Bsh[64 * LROW];   // 66560 B
  int lane = threadIdx.x & 31;
  int wv   = threadIdx.x >> 5;
  int half = lane >> 4;
  int l16  = lane & 15;

  int blk = blockIdx.x;          // 4608 blocks = 16 b * 8 nt * 36
  int b   = blk / 288;
  int r2  = blk % 288;
  int nt  = r2 / 36;             // uniform per block
  int mt  = (r2 % 36) * 8 + wv;  // 0..287

  if (threadIdx.x == 0)
    tdm_load_1d_padded(lds_off(Bsh), vhT + (size_t)nt * 64 * R_, 64 * 512);

  const bf16_t* Arow = a_bf + ((size_t)b * MDIM + mt * 16 + l16) * R_;
  const bf16_t* Bl[4];
#pragma unroll
  for (int t = 0; t < 4; ++t)
    Bl[t] = Bsh + (size_t)(t * 16 + l16) * LROW + half * 16;

  AFrag ac, an;
  ac.h[0] = *(const v8bf*)(Arow + half * 16);
  ac.h[1] = *(const v8bf*)(Arow + 8 + half * 16);
  an = ac;

  if (threadIdx.x == 0) __builtin_amdgcn_s_wait_tensorcnt(0);
  __syncthreads();

  v8f acc[4] = {{}, {}, {}, {}};
  for (int kb = 0; kb < 16; ++kb) {
    int k0 = kb * 32;
    if (kb < 15) {   // prefetch next A fragment while WMMAs consume current
      an.h[0] = *(const v8bf*)(Arow + k0 + 32 + half * 16);
      an.h[1] = *(const v8bf*)(Arow + k0 + 40 + half * 16);
    }
#pragma unroll
    for (int t = 0; t < 4; ++t) {
      AFrag bf;
      bf.h[0] = *(const v8bf*)(Bl[t] + k0);
      bf.h[1] = *(const v8bf*)(Bl[t] + k0 + 8);
      acc[t] = __builtin_amdgcn_wmma_f32_16x16x32_bf16(
          false, ac.v, false, bf.v, (short)0, acc[t], false, false);
    }
    ac = an;
  }

  float ss = 0.f;
#pragma unroll
  for (int t = 0; t < 4; ++t)
#pragma unroll
    for (int i = 0; i < 8; ++i) ss += acc[t][i] * acc[t][i];
  for (int off = 16; off > 0; off >>= 1) ss += __shfl_xor(ss, off, 32);
  if (lane == 0) atomicAdd(&sumsq[b], ss);

#pragma unroll
  for (int t = 0; t < 4; ++t)
#pragma unroll
    for (int i = 0; i < 8; ++i) {
      int m = mt * 16 + i + 8 * half;
      int n = nt * 64 + t * 16 + l16;
      delta_bf[((size_t)b * MDIM + m) * COUT + n] = (bf16_t)acc[t][i];
    }
}

// ---------------------------------------------------------------------------
// K4a: demod[b][co] = rsqrt( sum (SCALE*s*(w + (shift/norm)*delta))^2 + 1e-8 )
// ---------------------------------------------------------------------------
__global__ void __launch_bounds__(256) k_demod(
    const float* __restrict__ weight,
    const bf16_t* __restrict__ delta_bf,
    const float* __restrict__ s,
    const float* __restrict__ sumsq,
    const float* __restrict__ shifts,
    float* __restrict__ demod) {
  int b  = blockIdx.x >> 9;
  int co = blockIdx.x & 511;
  float nrm = fmaxf(sqrtf(sumsq[b]), 1e-12f);
  float sd  = shifts[b] / nrm;

  float acc = 0.f;
  for (int idx = threadIdx.x; idx < CIN * 9; idx += 256) {
    int cin = idx / 9, r = idx % 9;
    float d = (float)delta_bf[((size_t)b * MDIM + r * CIN + cin) * COUT + co];
    float v = SCALE_F * s[b * CIN + cin] * (weight[(size_t)co * MDIM + idx] + sd * d);
    acc += v * v;
  }
  __shared__ float red[8];
  for (int off = 16; off > 0; off >>= 1) acc += __shfl_xor(acc, off, 32);
  if ((threadIdx.x & 31) == 0) red[threadIdx.x >> 5] = acc;
  __syncthreads();
  if (threadIdx.x == 0) {
    float tot = 0.f;
    for (int i = 0; i < 8; ++i) tot += red[i];
    demod[b * COUT + co] = rsqrtf(tot + 1e-8f);
  }
}

// ---------------------------------------------------------------------------
// K4b: wkk[b][r][co][cin] = bf16( demod * SCALE * s * (w + (shift/norm)*delta) )
// ---------------------------------------------------------------------------
__global__ void k_pack_w(const float* __restrict__ weight,
                         const bf16_t* __restrict__ delta_bf,
                         const float* __restrict__ s,
                         const float* __restrict__ sumsq,
                         const float* __restrict__ shifts,
                         const float* __restrict__ demod,
                         bf16_t* __restrict__ wkk) {
  size_t idx = (size_t)blockIdx.x * 256 + threadIdx.x;   // B*9*COUT*CIN
  int cin = (int)(idx & 511);
  size_t t = idx >> 9;
  int co = (int)(t & 511);
  t >>= 9;
  int r = (int)(t % 9);
  int b = (int)(t / 9);

  float nrm = fmaxf(sqrtf(sumsq[b]), 1e-12f);
  float sd  = shifts[b] / nrm;
  float d   = (float)delta_bf[((size_t)b * MDIM + r * CIN + cin) * COUT + co];
  float wv  = weight[((size_t)co * CIN + cin) * 9 + r];
  float v   = demod[b * COUT + co] * SCALE_F * s[b * CIN + cin] * (wv + sd * d);
  wkk[idx] = (bf16_t)v;
}

// ---------------------------------------------------------------------------
// K5: conv as 9 shifted GEMMs. Block: 8 waves sharing (b, mt); per tap r the
//     16x512 weight panel is TDM-staged into double-buffered padded LDS and
//     pipelined across taps. Border zeros via per-lane predicated B loads.
// ---------------------------------------------------------------------------
__global__ void __launch_bounds__(256) k_conv(
    const bf16_t* __restrict__ xT,
    const bf16_t* __restrict__ wkk,
    float* __restrict__ out) {
  __shared__ __align__(16) bf16_t Ash[2][16 * LROW];   // 2 x 16640 B
  int lane = threadIdx.x & 31;
  int wv   = threadIdx.x >> 5;
  int half = lane >> 4;
  int l16  = lane & 15;

  int blk = blockIdx.x;            // 1024 blocks
  int b   = blk >> 6;              // 64 blocks per batch
  int rem = blk & 63;
  int mt  = rem >> 1;              // uniform per block: cout tile
  int pt  = (rem & 1) * 8 + wv;    // 0..15 : 64-wide spatial tile

  const bf16_t* Abase = wkk + ((size_t)b * 9 * COUT + mt * 16) * CIN;

  if (threadIdx.x == 0)
    tdm_load_1d_padded(lds_off(&Ash[0][0]), Abase, 16 * 512);

  int p[4], hc[4], wc[4];
#pragma unroll
  for (int t = 0; t < 4; ++t) {
    p[t]  = pt * 64 + t * 16 + l16;
    hc[t] = p[t] >> 5;
    wc[t] = p[t] & 31;
  }

  v8f acc[4] = {{}, {}, {}, {}};
  v8bf z8 = {};

  for (int r = 0; r < 9; ++r) {
    int cur = r & 1;
    if (threadIdx.x == 0) __builtin_amdgcn_s_wait_tensorcnt(0);
    __syncthreads();
    if (r + 1 < 9 && threadIdx.x == 0)
      tdm_load_1d_padded(lds_off(&Ash[cur ^ 1][0]),
                         Abase + (size_t)(r + 1) * COUT * CIN, 16 * 512);

    int dh = r / 3 - 1, dw = r % 3 - 1;
    const bf16_t* Bp[4];
    bool valid[4];
#pragma unroll
    for (int t = 0; t < 4; ++t) {
      int hh = hc[t] + dh, ww = wc[t] + dw;
      valid[t] = (hh >= 0) && (hh < 32) && (ww >= 0) && (ww < 32);
      Bp[t] = xT + ((size_t)b * HW_ + hh * 32 + ww) * CIN + half * 16;
    }
    const bf16_t* Al = &Ash[cur][0] + (size_t)l16 * LROW;

    for (int k0 = 0; k0 < CIN; k0 += 32) {
      AFrag af;
      af.h[0] = *(const v8bf*)(Al + k0 + half * 16);
      af.h[1] = *(const v8bf*)(Al + k0 + 8 + half * 16);
#pragma unroll
      for (int t = 0; t < 4; ++t) {
        AFrag bv;
        if (valid[t]) {
          bv.h[0] = *(const v8bf*)(Bp[t] + k0);
          bv.h[1] = *(const v8bf*)(Bp[t] + k0 + 8);
        } else {
          bv.h[0] = z8;
          bv.h[1] = z8;
        }
        acc[t] = __builtin_amdgcn_wmma_f32_16x16x32_bf16(
            false, af.v, false, bv.v, (short)0, acc[t], false, false);
      }
    }
    __syncthreads();
  }

#pragma unroll
  for (int t = 0; t < 4; ++t)
#pragma unroll
    for (int i = 0; i < 8; ++i) {
      int co = mt * 16 + i + 8 * half;
      out[((size_t)b * COUT + co) * HW_ + p[t]] = acc[t][i];
    }
}

// ---------------------------------------------------------------------------
extern "C" void kernel_launch(void* const* d_in, const int* in_sizes, int n_in,
                              void* d_out, int out_size, void* d_ws, size_t ws_size,
                              hipStream_t stream) {
  const float* x         = (const float*)d_in[0];
  const float* style     = (const float*)d_in[1];
  const float* mw        = (const float*)d_in[2];
  const float* mb        = (const float*)d_in[3];
  const float* weight    = (const float*)d_in[4];
  const float* u         = (const float*)d_in[5];
  const float* vh        = (const float*)d_in[6];
  const float* dir_delta = (const float*)d_in[7];
  const float* shifts    = (const float*)d_in[8];
  const int*   bdir      = (const int*)d_in[9];
  float* out = (float*)d_out;

  // workspace layout (256B aligned); a_bf region is reused as wkk
  char* ws = (char*)d_ws;
  float*  s      = (float*)(ws + 0);            //  32768 B
  float*  sumsq  = (float*)(ws + 32768);        //    256 B
  float*  demod  = (float*)(ws + 33024);        //  32768 B
  bf16_t* vhT    = (bf16_t*)(ws + 65792);       // 524288 B
  bf16_t* xT     = (bf16_t*)(ws + 590080);      // 16777216 B
  bf16_t* a_bf   = (bf16_t*)(ws + 17367296);    // 75497472 B (later: wkk)
  bf16_t* delta  = (bf16_t*)(ws + 92864768);    // 75497472 B
  // total: 168362240 B

  k_modulation<<<B_, 256, 0, stream>>>(style, mw, mb, s, sumsq);
  k_pack_a   <<<(B_ * (size_t)MDIM * R_) / 256, 256, 0, stream>>>(u, dir_delta, bdir, a_bf);
  k_pack_vhT <<<(COUT * R_) / 256, 256, 0, stream>>>(vh, vhT);
  k_pack_xT  <<<(B_ * HW_ * CIN) / 256, 256, 0, stream>>>(x, xT);
  k_delta_gemm<<<B_ * 288, 256, 0, stream>>>(a_bf, vhT, delta, sumsq);
  k_demod    <<<B_ * COUT, 256, 0, stream>>>(weight, delta, s, sumsq, shifts, demod);
  k_pack_w   <<<(B_ * 9 * (size_t)COUT * CIN) / 256, 256, 0, stream>>>(
      weight, delta, s, sumsq, shifts, demod, a_bf);
  k_conv     <<<(B_ * 512) / 8, 256, 0, stream>>>(xT, a_bf, out);
}
